// SkyworkMoE_37555194036876
// MI455X (gfx1250) — compile-verified, matrix operand order checked by reading, not run on hardware
//
#include <hip/hip_runtime.h>
#include <hip/hip_bf16.h>

#define H_DIM 2048
#define I_DIM 2816
#define E_N   16
#define T_N   8192
#define KB    64          // K-step per barrier pair (2x WMMA per stage)
#define SA    72          // LDS row stride in bf16 elems (144B, 16B-aligned)

typedef __attribute__((ext_vector_type(16))) __bf16 v16bf;
typedef __attribute__((ext_vector_type(8)))  __bf16 v8bf;
typedef __attribute__((ext_vector_type(8)))  float  v8f;
typedef __attribute__((ext_vector_type(4)))  float  v4f;

static __device__ __forceinline__ int imin(int a, int b) { return a < b ? a : b; }

// convert 16 consecutive fp32 -> two v8bf
static __device__ __forceinline__ void cvt16(const float* __restrict__ src,
                                             v8bf& p0, v8bf& p1) {
  const v4f f0 = *(const v4f*)(src + 0);
  const v4f f1 = *(const v4f*)(src + 4);
  const v4f f2 = *(const v4f*)(src + 8);
  const v4f f3 = *(const v4f*)(src + 12);
#pragma unroll
  for (int i = 0; i < 4; ++i) {
    p0[i] = (__bf16)f0[i]; p0[4 + i] = (__bf16)f1[i];
    p1[i] = (__bf16)f2[i]; p1[4 + i] = (__bf16)f3[i];
  }
}

// ---------------- Router: logits = tanh(x@wg1.T)@wg2.T, softmax, top-2 ----
__global__ __launch_bounds__(128) void k_router(const float* __restrict__ x,
                                                const float* __restrict__ wg1,
                                                const float* __restrict__ wg2,
                                                int* __restrict__ tidx,
                                                float* __restrict__ tw) {
  __shared__ float xs[H_DIM];
  __shared__ float h1[128];
  __shared__ float lg[16];
  const int t = blockIdx.x;
  const int j = threadIdx.x;
  for (int h = j; h < H_DIM; h += 128) xs[h] = x[(size_t)t * H_DIM + h];
  __syncthreads();
  float acc = 0.f;
  const float* wr = wg1 + (size_t)j * H_DIM;
  for (int h = 0; h < H_DIM; h += 4) {
    acc += xs[h] * wr[h] + xs[h + 1] * wr[h + 1] +
           xs[h + 2] * wr[h + 2] + xs[h + 3] * wr[h + 3];
  }
  h1[j] = tanhf(acc);
  __syncthreads();
  if (j < 16) {
    float a = 0.f;
    const float* w2 = wg2 + j * 128;
    for (int k = 0; k < 128; ++k) a += h1[k] * w2[k];
    lg[j] = a;
  }
  __syncthreads();
  if (j == 0) {
    float mx = lg[0];
    for (int e = 1; e < 16; ++e) mx = fmaxf(mx, lg[e]);
    float p[16], s = 0.f;
    for (int e = 0; e < 16; ++e) { p[e] = __expf(lg[e] - mx); s += p[e]; }
    const float inv = 1.f / s;
    for (int e = 0; e < 16; ++e) p[e] *= inv;
    int i0 = 0; float p0 = p[0];
    for (int e = 1; e < 16; ++e) if (p[e] > p0) { p0 = p[e]; i0 = e; }
    int i1 = (i0 == 0) ? 1 : 0; float p1 = p[i1];
    for (int e = 0; e < 16; ++e) if (e != i0 && p[e] > p1) { p1 = p[e]; i1 = e; }
    const float rn = 1.f / (p0 + p1);
    tidx[2 * t + 0] = i0; tidx[2 * t + 1] = i1;
    tw[2 * t + 0] = p0 * rn; tw[2 * t + 1] = p1 * rn;
  }
}

// ---------------- Per-expert token gathering ------------------------------
__global__ void k_count(const int* __restrict__ tidx, int* __restrict__ counts) {
  int t = blockIdx.x * blockDim.x + threadIdx.x;
  if (t < T_N) {
    atomicAdd(&counts[tidx[2 * t + 0]], 1);
    atomicAdd(&counts[tidx[2 * t + 1]], 1);
  }
}

__global__ void k_prefix(const int* __restrict__ counts, int* __restrict__ offsets) {
  if (threadIdx.x == 0) {
    int s = 0;
    for (int e = 0; e < E_N; ++e) { offsets[e] = s; s += counts[e]; }
  }
}

__global__ void k_scatter(const int* __restrict__ tidx, const float* __restrict__ tw,
                          const int* __restrict__ offsets, int* __restrict__ cursor,
                          int* __restrict__ rowmap, float* __restrict__ coefmap) {
  int t = blockIdx.x * blockDim.x + threadIdx.x;
  if (t < T_N) {
    for (int k = 0; k < 2; ++k) {
      const int e = tidx[2 * t + k];
      const int slot = atomicAdd(&cursor[e], 1);
      const int cr = offsets[e] + slot;
      rowmap[cr] = t;
      coefmap[cr] = tw[2 * t + k];
    }
  }
}

// ---------------- GEMM1: g,u = X@Wup.T ; h = silu(g)*u (bf16 WMMA) --------
// Block tile: M=128 gathered tokens, N=32 intermediate cols, K step 64.
// 8 waves: waves 0-3 gate (M sub-tiles), waves 4-7 up. 8 WMMAs per wave/stage.
__global__ __launch_bounds__(256) void k_gemm1(const float* __restrict__ x,
                                               const float* __restrict__ w_up,
                                               const int* __restrict__ counts,
                                               const int* __restrict__ offsets,
                                               const int* __restrict__ rowmap,
                                               __bf16* __restrict__ hbuf) {
  __shared__ __bf16 As[128 * SA];
  __shared__ __bf16 Bs[64 * SA];
  __shared__ float  gbuf[128 * 33];
  __shared__ float  ubuf[128 * 33];

  const int e = blockIdx.z;
  const int cnt = counts[e];
  const int rowbase = blockIdx.y * 128;
  if (rowbase >= cnt) return;
  const int rows_in = imin(128, cnt - rowbase);
  const int off = offsets[e];
  const int nbase = blockIdx.x * 32;

  const int tid = threadIdx.x;
  const int lane = tid & 31;
  const int w = tid >> 5;
  const int is_up = w >> 2;
  const int wm = (w & 3) * 32;

  // A-load assignment: 2 threads per row, 32 floats each
  const int arow = tid >> 1;
  const int akh = (tid & 1) * 32;
  const bool arow_ok = arow < rows_in;
  int tok = 0;
  if (arow_ok) tok = rowmap[off + rowbase + arow];
  const float* xrow = x + (size_t)tok * H_DIM;

  // B-load assignment (threads 0..127): 64 weight rows (32 gate + 32 up)
  const int brow = tid >> 1;
  const int bkh = (tid & 1) * 32;
  const float* wrow = nullptr;
  if (tid < 128) {
    const int n = nbase + (brow & 31);
    const size_t wri = (size_t)e * 2 * I_DIM + ((brow < 32) ? n : (I_DIM + n));
    wrow = w_up + wri * H_DIM;
  }

  v8f acc[2][2];
#pragma unroll
  for (int si = 0; si < 2; ++si)
#pragma unroll
    for (int sj = 0; sj < 2; ++sj)
      acc[si][sj] = (v8f){0.f, 0.f, 0.f, 0.f, 0.f, 0.f, 0.f, 0.f};

  for (int kk = 0; kk < H_DIM; kk += KB) {
    __syncthreads();
    // stage A (fp32 -> bf16): 32 elems per thread
    if (arow_ok) {
#pragma unroll
      for (int c = 0; c < 32; c += 16) {
        v8bf p0, p1;
        cvt16(&xrow[kk + akh + c], p0, p1);
        *(v8bf*)&As[arow * SA + akh + c + 0] = p0;
        *(v8bf*)&As[arow * SA + akh + c + 8] = p1;
      }
      if (kk + KB < H_DIM) __builtin_prefetch(&xrow[kk + KB + akh], 0, 1);
    } else {
      v8bf z;
#pragma unroll
      for (int i = 0; i < 8; ++i) z[i] = (__bf16)0.f;
#pragma unroll
      for (int c = 0; c < 32; c += 8) *(v8bf*)&As[arow * SA + akh + c] = z;
    }
    // stage B (fp32 -> bf16): 32 elems per thread (threads 0..127)
    if (tid < 128) {
#pragma unroll
      for (int c = 0; c < 32; c += 16) {
        v8bf q0, q1;
        cvt16(&wrow[kk + bkh + c], q0, q1);
        *(v8bf*)&Bs[brow * SA + bkh + c + 0] = q0;
        *(v8bf*)&Bs[brow * SA + bkh + c + 8] = q1;
      }
      if (kk + KB < H_DIM) __builtin_prefetch(&wrow[kk + KB + bkh], 0, 1);
    }
    __syncthreads();
    // fragments + 8 WMMAs per wave per stage
#pragma unroll
    for (int kb = 0; kb < KB; kb += 32) {
      const int ko = kb + (lane >> 4) * 8;
#pragma unroll
      for (int si = 0; si < 2; ++si) {
        union { v16bf v; v8bf h[2]; } a;
        const int m = wm + si * 16 + (lane & 15);
        a.h[0] = *(const v8bf*)&As[m * SA + ko];
        a.h[1] = *(const v8bf*)&As[m * SA + ko + 16];
#pragma unroll
        for (int sj = 0; sj < 2; ++sj) {
          union { v16bf v; v8bf h[2]; } b;
          const int n = is_up * 32 + sj * 16 + (lane & 15);
          b.h[0] = *(const v8bf*)&Bs[n * SA + ko];
          b.h[1] = *(const v8bf*)&Bs[n * SA + ko + 16];
          acc[si][sj] = __builtin_amdgcn_wmma_f32_16x16x32_bf16(
              false, a.v, false, b.v, (short)0, acc[si][sj], false, false);
        }
      }
    }
  }

  // spill accumulators (gate -> gbuf, up -> ubuf)
  float* obuf = is_up ? ubuf : gbuf;
#pragma unroll
  for (int si = 0; si < 2; ++si)
#pragma unroll
    for (int sj = 0; sj < 2; ++sj)
#pragma unroll
      for (int r = 0; r < 8; ++r) {
        const int m = wm + si * 16 + ((lane >> 4) << 3) + r;
        const int n = sj * 16 + (lane & 15);
        obuf[m * 33 + n] = acc[si][sj][r];
      }
  __syncthreads();

  // h = silu(g) * u -> bf16 workspace (2 threads per row, 16 cols each)
  {
    const int row = tid >> 1;
    const int kh = (tid & 1) * 16;
    if (row < rows_in) {
      v8bf h0, h1;
#pragma unroll
      for (int i = 0; i < 16; ++i) {
        const float g = gbuf[row * 33 + kh + i];
        const float u = ubuf[row * 33 + kh + i];
        const float s = g / (1.f + __expf(-g));
        const __bf16 hv = (__bf16)(s * u);
        if (i < 8) h0[i] = hv; else h1[i - 8] = hv;
      }
      __bf16* dst = hbuf + (size_t)(off + rowbase + row) * I_DIM + nbase + kh;
      *(v8bf*)(dst + 0) = h0;
      *(v8bf*)(dst + 8) = h1;
    }
  }
}

// ---------------- GEMM2: out[token] += coef * h @ Wdown.T -----------------
// Block tile: M=128 rows, N=64 hidden cols, K step 64 over I.
__global__ __launch_bounds__(256) void k_gemm2(const __bf16* __restrict__ hbuf,
                                               const float* __restrict__ w_down,
                                               const int* __restrict__ counts,
                                               const int* __restrict__ offsets,
                                               const int* __restrict__ rowmap,
                                               const float* __restrict__ coefmap,
                                               float* __restrict__ out) {
  __shared__ __bf16 As[128 * SA];
  __shared__ __bf16 Bs[64 * SA];
  __shared__ int    toks[128];
  __shared__ float  cofs[128];

  const int e = blockIdx.z;
  const int cnt = counts[e];
  const int rowbase = blockIdx.y * 128;
  if (rowbase >= cnt) return;
  const int rows_in = imin(128, cnt - rowbase);
  const int off = offsets[e];
  const int nbase = blockIdx.x * 64;

  const int tid = threadIdx.x;
  const int lane = tid & 31;
  const int w = tid >> 5;
  const int wm = (w & 3) * 32;
  const int wn = (w >> 2) * 32;

  if (tid < 128) {
    if (tid < rows_in) {
      const int cr = off + rowbase + tid;
      toks[tid] = rowmap[cr];
      cofs[tid] = coefmap[cr];
    } else { toks[tid] = 0; cofs[tid] = 0.f; }
  }

  const int arow = tid >> 1;
  const int akh = (tid & 1) * 32;
  int crow = off + rowbase + arow;
  if (crow > 2 * T_N - 1) crow = 2 * T_N - 1;  // clamp padded rows (discarded)
  const __bf16* hrow = hbuf + (size_t)crow * I_DIM;

  const int brow = tid >> 1;
  const int bkh = (tid & 1) * 32;
  const float* wrow = nullptr;
  if (tid < 128) wrow = w_down + ((size_t)e * H_DIM + nbase + brow) * I_DIM;

  v8f acc[2][2];
#pragma unroll
  for (int si = 0; si < 2; ++si)
#pragma unroll
    for (int sj = 0; sj < 2; ++sj)
      acc[si][sj] = (v8f){0.f, 0.f, 0.f, 0.f, 0.f, 0.f, 0.f, 0.f};

  for (int kk = 0; kk < I_DIM; kk += KB) {
    __syncthreads();
    // stage A: 32 bf16 per thread (already bf16)
    {
#pragma unroll
      for (int c = 0; c < 32; c += 8) {
        const v8bf a0 = *(const v8bf*)&hrow[kk + akh + c];
        *(v8bf*)&As[arow * SA + akh + c] = a0;
      }
      if (kk + KB < I_DIM) __builtin_prefetch(&hrow[kk + KB + akh], 0, 1);
    }
    // stage B: 32 fp32 -> bf16 per thread (threads 0..127)
    if (tid < 128) {
#pragma unroll
      for (int c = 0; c < 32; c += 16) {
        v8bf q0, q1;
        cvt16(&wrow[kk + bkh + c], q0, q1);
        *(v8bf*)&Bs[brow * SA + bkh + c + 0] = q0;
        *(v8bf*)&Bs[brow * SA + bkh + c + 8] = q1;
      }
      if (kk + KB < I_DIM) __builtin_prefetch(&wrow[kk + KB + bkh], 0, 1);
    }
    __syncthreads();
#pragma unroll
    for (int kb = 0; kb < KB; kb += 32) {
      const int ko = kb + (lane >> 4) * 8;
#pragma unroll
      for (int si = 0; si < 2; ++si) {
        union { v16bf v; v8bf h[2]; } a;
        const int m = wm + si * 16 + (lane & 15);
        a.h[0] = *(const v8bf*)&As[m * SA + ko];
        a.h[1] = *(const v8bf*)&As[m * SA + ko + 16];
#pragma unroll
        for (int sj = 0; sj < 2; ++sj) {
          union { v16bf v; v8bf h[2]; } b;
          const int n = wn + sj * 16 + (lane & 15);
          b.h[0] = *(const v8bf*)&Bs[n * SA + ko];
          b.h[1] = *(const v8bf*)&Bs[n * SA + ko + 16];
          acc[si][sj] = __builtin_amdgcn_wmma_f32_16x16x32_bf16(
              false, a.v, false, b.v, (short)0, acc[si][sj], false, false);
        }
      }
    }
  }

  // scaled scatter-add: each out element receives exactly 2 commutative adds
#pragma unroll
  for (int si = 0; si < 2; ++si)
#pragma unroll
    for (int sj = 0; sj < 2; ++sj)
#pragma unroll
      for (int r = 0; r < 8; ++r) {
        const int m = wm + si * 16 + ((lane >> 4) << 3) + r;
        if (m < rows_in) {
          const int n = wn + sj * 16 + (lane & 15);
          const float v = cofs[m] * acc[si][sj][r];
          unsafeAtomicAdd(&out[(size_t)toks[m] * H_DIM + nbase + n], v);
        }
      }
}

// ---------------- host-side launcher --------------------------------------
extern "C" void kernel_launch(void* const* d_in, const int* in_sizes, int n_in,
                              void* d_out, int out_size, void* d_ws, size_t ws_size,
                              hipStream_t stream) {
  (void)in_sizes; (void)n_in; (void)ws_size;
  const float* x      = (const float*)d_in[0];
  const float* wg1    = (const float*)d_in[1];
  const float* wg2    = (const float*)d_in[2];
  const float* w_up   = (const float*)d_in[3];
  const float* w_down = (const float*)d_in[4];
  float* out = (float*)d_out;

  char* ws = (char*)d_ws;
  int*    counts  = (int*)(ws + 0);
  int*    cursor  = (int*)(ws + 64);
  int*    offsets = (int*)(ws + 128);
  int*    tidx    = (int*)(ws + 256);
  float*  tw      = (float*)(ws + 256 + 1 * (size_t)2 * T_N * 4);
  int*    rowmap  = (int*)(ws + 256 + 2 * (size_t)2 * T_N * 4);
  float*  coefmap = (float*)(ws + 256 + 3 * (size_t)2 * T_N * 4);
  __bf16* hbuf    = (__bf16*)(ws + 512 + 4 * (size_t)2 * T_N * 4);

  hipMemsetAsync(ws, 0, 256, stream);                               // counts/cursor/offsets
  hipMemsetAsync(d_out, 0, (size_t)out_size * sizeof(float), stream);

  k_router<<<T_N, 128, 0, stream>>>(x, wg1, wg2, tidx, tw);
  k_count<<<T_N / 256, 256, 0, stream>>>(tidx, counts);
  k_prefix<<<1, 32, 0, stream>>>(counts, offsets);
  k_scatter<<<T_N / 256, 256, 0, stream>>>(tidx, tw, offsets, cursor, rowmap, coefmap);

  dim3 g1(I_DIM / 32, T_N / 128, E_N);   // 88 x 64 x 16 (early-exit past counts)
  k_gemm1<<<g1, 256, 0, stream>>>(x, w_up, counts, offsets, rowmap, hbuf);

  dim3 g2(H_DIM / 64, T_N / 128, E_N);   // 32 x 64 x 16
  k_gemm2<<<g2, 256, 0, stream>>>(hbuf, w_down, counts, offsets, rowmap, coefmap, out);
}
